// MambaBlock_22548578304658
// MI455X (gfx1250) — matrix operand order repaired and checked by reading, not compiled, not run on hardware
//
#include <hip/hip_runtime.h>

// ---------------------------------------------------------------------------
// Mamba block forward for MI455X (gfx1250, wave32, WMMA).
// B=2, T=2048, D_MODEL=1024, D_INNER=2048, DT_RANK=64, D_STATE=16, D_CONV=4.
//  * GEMMs: v_wmma_f32_16x16x32_bf16, fp32 accumulate. Both A and B operands
//    are pre-packed into lane-contiguous fragment layouts so every WMMA
//    operand load is a coalesced global_load_b128 pair. Wave tile = 64x64
//    (16 accumulators) -> 1.0 b128 load per WMMA.
//  * SSM scan: chunked linear-recurrence scan (3 passes) instead of a
//    2048-step serial chain -> sequential depth ~64+32+64.
// ---------------------------------------------------------------------------

#define BB     2
#define TT     2048
#define DMODEL 1024
#define DINNER 2048
#define DTRANK 64
#define DSTATE 16
#define DCONV  4
#define MROWS  (BB * TT)          // 4096
#define CHUNK  64                 // scan chunk length
#define NCB    (TT / CHUNK)       // 32 chunks per batch
#define NCHUNK (BB * NCB)         // 64 chunks total

typedef __attribute__((ext_vector_type(16))) __bf16 v16bf;
typedef __attribute__((ext_vector_type(8)))  float  v8f;
typedef __attribute__((ext_vector_type(4)))  unsigned int u32x4;

union Frag16 { v16bf v; u32x4 q[2]; };

__device__ __forceinline__ __bf16 to_bf16(float f) {
  unsigned u = __builtin_bit_cast(unsigned, f);
  unsigned r = (u + 0x7FFFu + ((u >> 16) & 1u)) >> 16;   // round-nearest-even
  unsigned short s = (unsigned short)r;
  return __builtin_bit_cast(__bf16, s);
}

__device__ __forceinline__ float silu_f(float v) {
  return v / (1.0f + __expf(-v));
}

__device__ __forceinline__ float softplus_f(float v) {
  return (v > 20.0f) ? v : __logf(1.0f + __expf(v));
}

// ---------------------------------------------------------------------------
// Packed-A addressing. A bf16 M x K matrix is stored as 16x32 WMMA tiles;
// within a tile, lane L's 16 fragment values are contiguous:
//   elem_off = (tileM * (K/32) + tileK) * 512 + lane*16 + j
// with (per ISA A-layout): kk in [0,8)->h=0,j=kk ; [8,16)->h=1,j=kk-8 ;
// [16,24)->h=0,j=kk-8 ; [24,32)->h=1,j=kk-16 ; lane = (row&15) + 16*h.
// ---------------------------------------------------------------------------
__device__ __forceinline__ size_t packA_off(int row, int k, int K) {
  const int kk   = k & 31;
  const int h    = (kk >> 3) & 1;
  const int j    = (kk & 7) | ((kk >> 4) << 3);
  const int lane = (row & 15) | (h << 4);
  return ((size_t)(row >> 4) * (K >> 5) + (k >> 5)) * 512 + lane * 16 + j;
}

// ---------------------------------------------------------------------------
// Pack an M x K fp32 row-major matrix into packed-A bf16 (used for x).
// ---------------------------------------------------------------------------
__global__ void pack_a_f32_kernel(const float* __restrict__ src,
                                  __bf16* __restrict__ dst, int M, int K) {
  size_t i = (size_t)blockIdx.x * blockDim.x + threadIdx.x;
  if (i >= (size_t)M * K) return;
  int row = (int)(i / K);
  int k   = (int)(i - (size_t)row * K);
  dst[packA_off(row, k, K)] = to_bf16(src[i]);
}

// ---------------------------------------------------------------------------
// Pack a KxN fp32 weight into bf16 WMMA-B layout: Wp[(k>>5)][n][k&31].
// ---------------------------------------------------------------------------
__global__ void pack_w_bf16_kernel(const float* __restrict__ W,
                                   __bf16* __restrict__ Wp, int K, int N) {
  int i = blockIdx.x * blockDim.x + threadIdx.x;
  if (i >= K * N) return;
  int k = i / N;
  int n = i - k * N;
  Wp[((size_t)(k >> 5) * N + n) * 32 + (k & 31)] = to_bf16(W[i]);
}

// ---------------------------------------------------------------------------
// Extract dt columns (0..63) of x_dbl into packed-A bf16 (4096 x 64).
// ---------------------------------------------------------------------------
__global__ void extract_dt_kernel(const float* __restrict__ xdbl,
                                  __bf16* __restrict__ dtb) {
  int i = blockIdx.x * blockDim.x + threadIdx.x;   // over 4096*64
  if (i >= MROWS * DTRANK) return;
  int row = i >> 6;
  int j   = i & 63;
  dtb[packA_off(row, j, DTRANK)] =
      to_bf16(xdbl[(size_t)row * (DTRANK + 2 * DSTATE) + j]);
}

// ---------------------------------------------------------------------------
// Main WMMA GEMM: C[M x ldc] = Ap(packed bf16, M x K) * Bp(packed bf16).
// Block = 256 threads = 8 waves arranged 2(M) x 4(N); wave tile = 64 x 64
// (4 M-frags x 4 N-frags, 16 v8f accumulators). Block tile = 128 x 256.
// EPI==1 adds bias[col] then softplus (delta path).
// ---------------------------------------------------------------------------
template <int EPI>
__global__ void wmma_gemm_bf16(const __bf16* __restrict__ Ap,
                               const __bf16* __restrict__ Bp,
                               float* __restrict__ C,
                               const float* __restrict__ bias,
                               int M, int N, int K, int ldc) {
  const int lane = threadIdx.x & 31;
  const int wave = threadIdx.x >> 5;
  const int wm   = wave >> 2;           // 0..1
  const int wn   = wave & 3;            // 0..3
  const int h    = lane >> 4;
  const int ln   = lane & 15;
  const int mBase = blockIdx.y * 128 + wm * 64;
  const int nBase = blockIdx.x * 256 + wn * 64;
  const int KT    = K >> 5;

  v8f acc[4][4];
#pragma unroll
  for (int mt = 0; mt < 4; ++mt)
#pragma unroll
    for (int nt = 0; nt < 4; ++nt) acc[mt][nt] = (v8f)0.0f;

  for (int kt = 0; kt < KT; ++kt) {
    Frag16 a[4];
#pragma unroll
    for (int mt = 0; mt < 4; ++mt) {
      // packed-A: whole wave's fragment = 1KB contiguous block
      const __bf16* ap =
          Ap + ((size_t)((mBase >> 4) + mt) * KT + kt) * 512 + lane * 16;
      a[mt].q[0] = *reinterpret_cast<const u32x4*>(ap);
      a[mt].q[1] = *reinterpret_cast<const u32x4*>(ap + 8);
    }
#pragma unroll
    for (int nt = 0; nt < 4; ++nt) {
      Frag16 b;
      const __bf16* bp =
          Bp + ((size_t)kt * N + (nBase + nt * 16 + ln)) * 32 + 8 * h;
      b.q[0] = *reinterpret_cast<const u32x4*>(bp);
      b.q[1] = *reinterpret_cast<const u32x4*>(bp + 16);
#pragma unroll
      for (int mt = 0; mt < 4; ++mt) {
        acc[mt][nt] = __builtin_amdgcn_wmma_f32_16x16x32_bf16(
            false, a[mt].v, false, b.v, (short)0, acc[mt][nt], false, false);
      }
    }
  }

  // Epilogue: C/D layout -> element r: row = r + 8h, col = ln.
#pragma unroll
  for (int mt = 0; mt < 4; ++mt) {
#pragma unroll
    for (int nt = 0; nt < 4; ++nt) {
      const int col = nBase + nt * 16 + ln;
#pragma unroll
      for (int r = 0; r < 8; ++r) {
        const int row = mBase + mt * 16 + r + 8 * h;
        float v = acc[mt][nt][r];
        if (EPI == 1) v = softplus_f(v + bias[col]);
        C[(size_t)row * ldc + col] = v;
      }
    }
  }
}

// ---------------------------------------------------------------------------
// Narrow-N WMMA GEMM for x_dbl: N = 96 (6 N-tiles). Wave = 16 rows x 96 cols.
// ---------------------------------------------------------------------------
__global__ void wmma_gemm_n96(const __bf16* __restrict__ Ap,  // packed M x K
                              const __bf16* __restrict__ Bp,  // packed, N=96
                              float* __restrict__ C,          // M x 96
                              int M, int K) {
  const int N    = DTRANK + 2 * DSTATE;  // 96
  const int lane = threadIdx.x & 31;
  const int wave = threadIdx.x >> 5;
  const int h    = lane >> 4;
  const int ln   = lane & 15;
  const int mBase = blockIdx.x * 128 + wave * 16;
  const int KT    = K >> 5;

  v8f acc[6];
#pragma unroll
  for (int nt = 0; nt < 6; ++nt) acc[nt] = (v8f)0.0f;

  for (int kt = 0; kt < KT; ++kt) {
    Frag16 a;
    const __bf16* ap =
        Ap + ((size_t)(mBase >> 4) * KT + kt) * 512 + lane * 16;
    a.q[0] = *reinterpret_cast<const u32x4*>(ap);
    a.q[1] = *reinterpret_cast<const u32x4*>(ap + 8);
#pragma unroll
    for (int nt = 0; nt < 6; ++nt) {
      Frag16 b;
      const __bf16* bp =
          Bp + ((size_t)kt * N + (nt * 16 + ln)) * 32 + 8 * h;
      b.q[0] = *reinterpret_cast<const u32x4*>(bp);
      b.q[1] = *reinterpret_cast<const u32x4*>(bp + 16);
      acc[nt] = __builtin_amdgcn_wmma_f32_16x16x32_bf16(
          false, a.v, false, b.v, (short)0, acc[nt], false, false);
    }
  }

#pragma unroll
  for (int nt = 0; nt < 6; ++nt) {
    const int col = nt * 16 + ln;
#pragma unroll
    for (int r = 0; r < 8; ++r) {
      const int row = mBase + r + 8 * h;
      C[(size_t)row * N + col] = acc[nt][r];
    }
  }
}

// ---------------------------------------------------------------------------
// Causal depthwise conv (width 4) + bias + SiLU. Reads xi (cols 0..2047 of
// xz), writes xc (fp32 for scan) and xcb (packed-A bf16 for the x_dbl GEMM).
// ---------------------------------------------------------------------------
__global__ void conv_silu_kernel(const float* __restrict__ xz,
                                 const float* __restrict__ conv_w,
                                 const float* __restrict__ conv_b,
                                 float* __restrict__ xc,
                                 __bf16* __restrict__ xcb) {
  size_t idx = (size_t)blockIdx.x * blockDim.x + threadIdx.x;  // 4096*2048
  if (idx >= (size_t)MROWS * DINNER) return;
  int d = (int)(idx & (DINNER - 1));
  size_t row = idx >> 11;                 // DINNER = 2048 = 2^11
  int t = (int)(row & (TT - 1));
  float acc = conv_b[d];
#pragma unroll
  for (int j = 0; j < DCONV; ++j) {
    int tt = t - (DCONV - 1) + j;
    if (tt >= 0)
      acc += conv_w[d * DCONV + j] *
             xz[(row - (DCONV - 1) + j) * (2 * DINNER) + d];
  }
  float s = silu_f(acc);
  xc[idx] = s;
  xcb[packA_off((int)row, d, DINNER)] = to_bf16(s);
}

// ---------------------------------------------------------------------------
// Chunked SSM scan.  h_{t+1} = dA_t h_t + dBu_t is linear in h, so over a
// chunk: h_end = (prod dA) * h_start + q.
// Pass 1: per (chunk, d) compute P[n], q[n] in parallel (B via LDS).
// Pass 2: serial over 32 chunks/batch: propagate chunk-entry states Hs.
// Pass 3: per (chunk, d) replay chunk with true h0, compute y, fuse +xc*D and
//         *silu(z), store packed bf16 for the output GEMM.
// ---------------------------------------------------------------------------
__global__ void scan_pass1_kernel(const float* __restrict__ delta,
                                  const float* __restrict__ xc,
                                  const float* __restrict__ xdbl,
                                  const float* __restrict__ A_log,
                                  float* __restrict__ P,
                                  float* __restrict__ Q) {
  const int d  = blockIdx.x * blockDim.x + threadIdx.x;  // 0..2047
  const int bc = blockIdx.y;                             // 0..63
  const int b  = bc >> 5;
  const int c  = bc & (NCB - 1);
  __shared__ float Bs[CHUNK][DSTATE];                    // 4 KB

  for (int i = threadIdx.x; i < CHUNK * DSTATE; i += blockDim.x) {
    int t = i >> 4, n = i & 15;
    size_t row = (size_t)b * TT + c * CHUNK + t;
    Bs[t][n] = xdbl[row * (DTRANK + 2 * DSTATE) + DTRANK + n];
  }
  __syncthreads();

  float Am[DSTATE], Pr[DSTATE], Qr[DSTATE];
#pragma unroll
  for (int n = 0; n < DSTATE; ++n) {
    Am[n] = -__expf(A_log[(size_t)d * DSTATE + n]);
    Pr[n] = 1.0f;
    Qr[n] = 0.0f;
  }

#pragma unroll 2
  for (int t = 0; t < CHUNK; ++t) {
    const size_t row = (size_t)b * TT + c * CHUNK + t;
    const float dl = delta[row * DINNER + d];
    const float xv = xc[row * DINNER + d];
#pragma unroll
    for (int n = 0; n < DSTATE; ++n) {
      const float dA  = __expf(dl * Am[n]);
      const float dBu = dl * Bs[t][n] * xv;
      Pr[n] *= dA;
      Qr[n] = dA * Qr[n] + dBu;
    }
  }

  const size_t o = ((size_t)bc * DINNER + d) * DSTATE;
#pragma unroll
  for (int n = 0; n < DSTATE; ++n) {
    P[o + n] = Pr[n];
    Q[o + n] = Qr[n];
  }
}

__global__ void scan_pass2_kernel(const float* __restrict__ P,
                                  const float* __restrict__ Q,
                                  float* __restrict__ Hs) {
  const int d = blockIdx.x * blockDim.x + threadIdx.x;  // 0..2047
  const int b = blockIdx.y;
  float h[DSTATE];
#pragma unroll
  for (int n = 0; n < DSTATE; ++n) h[n] = 0.0f;
  for (int c = 0; c < NCB; ++c) {
    const size_t o = ((size_t)(b * NCB + c) * DINNER + d) * DSTATE;
#pragma unroll
    for (int n = 0; n < DSTATE; ++n) Hs[o + n] = h[n];
#pragma unroll
    for (int n = 0; n < DSTATE; ++n) h[n] = P[o + n] * h[n] + Q[o + n];
  }
}

__global__ void scan_pass3_kernel(const float* __restrict__ delta,
                                  const float* __restrict__ xc,
                                  const float* __restrict__ xdbl,
                                  const float* __restrict__ xz,   // for z
                                  const float* __restrict__ A_log,
                                  const float* __restrict__ Dvec,
                                  const float* __restrict__ Hs,
                                  __bf16* __restrict__ ygb) {
  const int d  = blockIdx.x * blockDim.x + threadIdx.x;  // 0..2047
  const int bc = blockIdx.y;                             // 0..63
  const int b  = bc >> 5;
  const int c  = bc & (NCB - 1);
  __shared__ float BCs[CHUNK][2 * DSTATE];               // 8 KB

  for (int i = threadIdx.x; i < CHUNK * 2 * DSTATE; i += blockDim.x) {
    int t = i >> 5, w = i & 31;
    size_t row = (size_t)b * TT + c * CHUNK + t;
    BCs[t][w] = xdbl[row * (DTRANK + 2 * DSTATE) + DTRANK + w];
  }
  __syncthreads();

  float Am[DSTATE], h[DSTATE];
  const size_t o = ((size_t)bc * DINNER + d) * DSTATE;
#pragma unroll
  for (int n = 0; n < DSTATE; ++n) {
    Am[n] = -__expf(A_log[(size_t)d * DSTATE + n]);
    h[n]  = Hs[o + n];
  }
  const float Dd = Dvec[d];

#pragma unroll 2
  for (int t = 0; t < CHUNK; ++t) {
    const size_t row = (size_t)b * TT + c * CHUNK + t;
    const float dl = delta[row * DINNER + d];
    const float xv = xc[row * DINNER + d];
    float y = 0.0f;
#pragma unroll
    for (int n = 0; n < DSTATE; ++n) {
      const float dA = __expf(dl * Am[n]);
      h[n] = dA * h[n] + dl * BCs[t][n] * xv;
      y += h[n] * BCs[t][DSTATE + n];
    }
    const float zv  = xz[row * (2 * DINNER) + DINNER + d];
    const float out = (y + xv * Dd) * silu_f(zv);
    ygb[packA_off((int)row, d, DINNER)] = to_bf16(out);
  }
}

// ---------------------------------------------------------------------------
// Host-side launch.
// ---------------------------------------------------------------------------
static size_t align_up(size_t x) { return (x + 255) & ~(size_t)255; }

extern "C" void kernel_launch(void* const* d_in, const int* in_sizes, int n_in,
                              void* d_out, int out_size, void* d_ws,
                              size_t ws_size, hipStream_t stream) {
  (void)in_sizes; (void)n_in; (void)out_size; (void)ws_size;

  const float* x      = (const float*)d_in[0];   // (B,T,1024)
  const float* W_in   = (const float*)d_in[1];   // (1024,4096)
  const float* conv_w = (const float*)d_in[2];   // (2048,4)
  const float* conv_b = (const float*)d_in[3];   // (2048,)
  const float* W_x    = (const float*)d_in[4];   // (2048,96)
  const float* W_dt   = (const float*)d_in[5];   // (64,2048)
  const float* b_dt   = (const float*)d_in[6];   // (2048,)
  const float* A_log  = (const float*)d_in[7];   // (2048,16)
  const float* Dvec   = (const float*)d_in[8];   // (2048,)
  const float* W_out  = (const float*)d_in[9];   // (2048,1024)
  float* out = (float*)d_out;                    // (B,T,1024)

  // --- workspace carve-out -------------------------------------------------
  char* base = (char*)d_ws;
  size_t off = 0;
  auto carve = [&](size_t bytes) {
    char* p = base + off;
    off = align_up(off + bytes);
    return p;
  };
  __bf16* xb     = (__bf16*)carve((size_t)MROWS * DMODEL * 2);       //  8 MB
  __bf16* W_inp  = (__bf16*)carve((size_t)DMODEL * 4096 * 2);        //  8 MB
  float*  xz     = (float*) carve((size_t)MROWS * 4096 * 4);         // 64 MB
  float*  xc     = (float*) carve((size_t)MROWS * DINNER * 4);       // 32 MB
  __bf16* xcb    = (__bf16*)carve((size_t)MROWS * DINNER * 2);       // 16 MB
  __bf16* W_xp   = (__bf16*)carve((size_t)DINNER * 96 * 2);
  float*  xdbl   = (float*) carve((size_t)MROWS * 96 * 4);
  __bf16* dtb    = (__bf16*)carve((size_t)MROWS * DTRANK * 2);
  __bf16* W_dtp  = (__bf16*)carve((size_t)DTRANK * DINNER * 2);
  float*  delta  = (float*) carve((size_t)MROWS * DINNER * 4);       // 32 MB
  __bf16* ygb    = (__bf16*)carve((size_t)MROWS * DINNER * 2);       // 16 MB
  __bf16* W_outp = (__bf16*)carve((size_t)DINNER * DMODEL * 2);      //  4 MB
  float*  Pc     = (float*) carve((size_t)NCHUNK * DINNER * DSTATE * 4); // 8MB
  float*  Qc     = (float*) carve((size_t)NCHUNK * DINNER * DSTATE * 4); // 8MB
  float*  Hs     = (float*) carve((size_t)NCHUNK * DINNER * DSTATE * 4); // 8MB

  const int TPB = 256;

  // --- 1. precision conversion / operand packing ---------------------------
  pack_a_f32_kernel<<<(MROWS * DMODEL + TPB - 1) / TPB, TPB, 0, stream>>>(
      x, xb, MROWS, DMODEL);
  pack_w_bf16_kernel<<<(DMODEL * 4096 + TPB - 1) / TPB, TPB, 0, stream>>>(
      W_in, W_inp, DMODEL, 4096);
  pack_w_bf16_kernel<<<(DINNER * 96 + TPB - 1) / TPB, TPB, 0, stream>>>(
      W_x, W_xp, DINNER, 96);
  pack_w_bf16_kernel<<<(DTRANK * DINNER + TPB - 1) / TPB, TPB, 0, stream>>>(
      W_dt, W_dtp, DTRANK, DINNER);
  pack_w_bf16_kernel<<<(DINNER * DMODEL + TPB - 1) / TPB, TPB, 0, stream>>>(
      W_out, W_outp, DINNER, DMODEL);

  // --- 2. xz = x @ W_in  (4096 x 4096 x 1024) ------------------------------
  wmma_gemm_bf16<0><<<dim3(4096 / 256, MROWS / 128), TPB, 0, stream>>>(
      xb, W_inp, xz, nullptr, MROWS, 4096, DMODEL, 4096);

  // --- 3. causal conv + SiLU -----------------------------------------------
  {
    size_t n = (size_t)MROWS * DINNER;
    conv_silu_kernel<<<(unsigned)((n + TPB - 1) / TPB), TPB, 0, stream>>>(
        xz, conv_w, conv_b, xc, xcb);
  }

  // --- 4. x_dbl = xc @ W_x  (4096 x 96 x 2048) -----------------------------
  wmma_gemm_n96<<<MROWS / 128, TPB, 0, stream>>>(xcb, W_xp, xdbl, MROWS,
                                                 DINNER);

  // --- 5. delta = softplus(dt @ W_dt + b_dt)  (4096 x 2048 x 64) -----------
  extract_dt_kernel<<<(MROWS * DTRANK + TPB - 1) / TPB, TPB, 0, stream>>>(
      xdbl, dtb);
  wmma_gemm_bf16<1><<<dim3(DINNER / 256, MROWS / 128), TPB, 0, stream>>>(
      dtb, W_dtp, delta, b_dt, MROWS, DINNER, DTRANK, DINNER);

  // --- 6. chunked SSM scan + gating ----------------------------------------
  scan_pass1_kernel<<<dim3(DINNER / TPB, NCHUNK), TPB, 0, stream>>>(
      delta, xc, xdbl, A_log, Pc, Qc);
  scan_pass2_kernel<<<dim3(DINNER / TPB, BB), TPB, 0, stream>>>(Pc, Qc, Hs);
  scan_pass3_kernel<<<dim3(DINNER / TPB, NCHUNK), TPB, 0, stream>>>(
      delta, xc, xdbl, xz, A_log, Dvec, Hs, ygb);

  // --- 7. out = y_gated @ W_out  (4096 x 1024 x 2048) ----------------------
  wmma_gemm_bf16<0><<<dim3(DMODEL / 256, MROWS / 128), TPB, 0, stream>>>(
      ygb, W_outp, out, nullptr, MROWS, DMODEL, DINNER, DMODEL);
}
